// SemanticAttention_8684423872853
// MI455X (gfx1250) — matrix operand order repaired and checked by reading, not compiled, not run on hardware
//
#include <hip/hip_runtime.h>
#include <hip/hip_bf16.h>

#define BN 8
#define NN 4096
#define CC 512
#define KK 150
#define KP 160   // K padded to multiple of 16

typedef __attribute__((ext_vector_type(16))) __bf16 bfx16;
typedef __attribute__((ext_vector_type(8)))  __bf16 bfx8;
typedef __attribute__((ext_vector_type(8)))  float  fx8;
typedef __attribute__((ext_vector_type(4)))  int    i4v;

#if defined(__has_builtin)
#if __has_builtin(__builtin_amdgcn_global_load_async_to_lds_b128)
#define HAVE_ASYNC_LDS 1
#endif
#endif

// ---------------- fragment helpers ----------------

// A-fragment (16x32 bf16, MxK): lane m=lane&15, kh=lane>>4.
// elements i=0..15 -> k = kh*16 + i (contiguous)
// caller passes p = row_base + kh*16 ; we fetch p[0..7] and p[8..15]
__device__ __forceinline__ bfx16 lda_frag(const __bf16* p) {
  bfx8 lo = *reinterpret_cast<const bfx8*>(p);
  bfx8 hi = *reinterpret_cast<const bfx8*>(p + 8);
  bfx16 r;
#pragma unroll
  for (int i = 0; i < 8; ++i) { r[i] = lo[i]; r[i + 8] = hi[i]; }
  return r;
}

// B-fragment (32x16 bf16, KxN): lane col=lane&15, kh=lane>>4.
// elements i=0..15 -> k = kh*16 + i (contiguous 32 bytes)
__device__ __forceinline__ bfx16 ldb_frag(const __bf16* p) {
  return *reinterpret_cast<const bfx16*>(p);
}

__device__ __forceinline__ fx8 wmma_bf16(bfx16 a, bfx16 b, fx8 c) {
  return __builtin_amdgcn_wmma_f32_16x16x32_bf16(
      false, a, false, b, (short)0, c, false, false);
}

__device__ __forceinline__ float redmax16(float v) {
#pragma unroll
  for (int m = 1; m < 16; m <<= 1) v = fmaxf(v, __shfl_xor(v, m, 32));
  return v;
}
__device__ __forceinline__ float redsum16(float v) {
#pragma unroll
  for (int m = 1; m < 16; m <<= 1) v += __shfl_xor(v, m, 32);
  return v;
}

// ---- async global -> LDS copy of 16 bytes (falls back to sync staging) ----
__device__ __forceinline__ void cp16(__bf16* dst_lds, const __bf16* src_glob) {
#ifdef HAVE_ASYNC_LDS
  __builtin_amdgcn_global_load_async_to_lds_b128(
      (__attribute__((address_space(1))) i4v*)src_glob,
      (__attribute__((address_space(3))) i4v*)dst_lds, 0, 0);
#else
  *reinterpret_cast<bfx8*>(dst_lds) = *reinterpret_cast<const bfx8*>(src_glob);
#endif
}

__device__ __forceinline__ void wait_async_all() {
#ifdef HAVE_ASYNC_LDS
#if __has_builtin(__builtin_amdgcn_s_wait_asynccnt)
  __builtin_amdgcn_s_wait_asynccnt(0);
#else
  asm volatile("s_wait_asynccnt 0" ::: "memory");
#endif
#endif
}

// ---------------- kernel 0: fp32 -> bf16 conversion + weight padding ----------------
__global__ void cvt_kernel(const float* __restrict__ x,
                           const float* __restrict__ Wq,
                           const float* __restrict__ Wk,
                           const float* __restrict__ Wv,
                           const float* __restrict__ Wres,
                           __bf16* __restrict__ xb,
                           __bf16* __restrict__ wqb,
                           __bf16* __restrict__ wkb,
                           __bf16* __restrict__ wvb,
                           __bf16* __restrict__ wrb) {
  const long total = (long)BN * NN * CC;
  long stride = (long)gridDim.x * blockDim.x;
  long tid = (long)blockIdx.x * blockDim.x + threadIdx.x;
  for (long i = tid; i < total; i += stride) xb[i] = (__bf16)x[i];
  for (long i = tid; i < (long)KP * CC; i += stride) {
    int j = (int)(i / CC), c = (int)(i % CC);
    wqb[i] = (j < KK) ? (__bf16)Wq[(long)j * CC + c] : (__bf16)0.0f;
    wkb[i] = (j < KK) ? (__bf16)Wk[(long)j * CC + c] : (__bf16)0.0f;
  }
  for (long i = tid; i < (long)CC * CC; i += stride) {
    wvb[i] = (__bf16)Wv[i];
    wrb[i] = (__bf16)Wres[i];
  }
}

// ---------------- kernel 1: Q/K projections ----------------
__global__ void proj_qk_kernel(const __bf16* __restrict__ xb,
                               const __bf16* __restrict__ wqb,
                               const __bf16* __restrict__ wkb,
                               const float* __restrict__ bq,
                               const float* __restrict__ bk,
                               float* __restrict__ segmap,
                               __bf16* __restrict__ Qb,
                               __bf16* __restrict__ Kb) {
  const int lane = threadIdx.x & 31;
  const int col = lane & 15, kh = lane >> 4;
  const int jbase = blockIdx.x << 4;
  const int rt = blockIdx.y;
  const int b = rt >> 8, n0 = (rt & 255) << 4;
  const int which = blockIdx.z;

  const __bf16* Wb = which ? wkb : wqb;
  const float* bias = which ? bk : bq;
  __bf16* Ob = which ? Kb : Qb;

  const __bf16* arow = xb + ((size_t)(b * NN + n0 + col)) * CC + kh * 16;
  const __bf16* brow = Wb + (size_t)(jbase + col) * CC + kh * 16;

  fx8 acc = {};
#pragma unroll 4
  for (int cc = 0; cc < CC; cc += 32)
    acc = wmma_bf16(lda_frag(arow + cc), ldb_frag(brow + cc), acc);

  const int jcol = jbase + col;
  const float bval = (jcol < KK) ? bias[jcol] : 0.0f;
#pragma unroll
  for (int j = 0; j < 8; ++j) {
    const int r = n0 + j + kh * 8;
    const float v = acc[j] + bval;
    Ob[((size_t)b * NN + r) * KP + jcol] = (jcol < KK) ? (__bf16)v : (__bf16)0.0f;
    if (which == 0 && jcol < KK)
      segmap[((size_t)b * NN + r) * KK + jcol] = v;
  }
}

// ---------------- kernel 2: V projection, stored transposed (b, C, N) ----------------
__global__ void proj_v_kernel(const __bf16* __restrict__ xb,
                              const __bf16* __restrict__ wvb,
                              const float* __restrict__ bv,
                              __bf16* __restrict__ Vt) {
  const int lane = threadIdx.x & 31;
  const int col = lane & 15, kh = lane >> 4;
  const int jbase = blockIdx.x << 4;
  const int rt = blockIdx.y;
  const int b = rt >> 8, n0 = (rt & 255) << 4;

  const __bf16* arow = xb + ((size_t)(b * NN + n0 + col)) * CC + kh * 16;
  const __bf16* brow = wvb + (size_t)(jbase + col) * CC + kh * 16;

  fx8 acc = {};
#pragma unroll 4
  for (int cc = 0; cc < CC; cc += 32)
    acc = wmma_bf16(lda_frag(arow + cc), ldb_frag(brow + cc), acc);

  const int ch = jbase + col;
  const float bval = bv[ch];
#pragma unroll
  for (int j = 0; j < 8; ++j) {
    const int r = n0 + j + kh * 8;
    Vt[((size_t)b * CC + ch) * NN + r] = (__bf16)(acc[j] + bval);
  }
}

// ---------------- kernel 3: flash attention ----------------
// Block = 4 waves; each wave owns a 16-row Q tile; the 4 waves share
// double-buffered K/V chunks staged in LDS with async global->LDS copies.
__global__ void __launch_bounds__(128, 1)
attn_kernel(const __bf16* __restrict__ Qb,
            const __bf16* __restrict__ Kb,
            const __bf16* __restrict__ Vt,
            __bf16* __restrict__ Fb) {
  __shared__ __align__(64) __bf16 ldsK[2][32 * KP];   // 2 x 10 KB
  __shared__ __align__(64) __bf16 ldsV[2][CC * 32];   // 2 x 32 KB
  __shared__ __align__(64) __bf16 pbuf[4][16 * 32];   // per-wave P tile

  const int tid = threadIdx.x;
  const int wave = tid >> 5;
  const int lane = tid & 31;
  const int col = lane & 15, kh = lane >> 4;
  const int rt = blockIdx.x;              // 0..511, 64 rows per block
  const int b = rt >> 6;
  const int n0 = ((rt & 63) << 6) + (wave << 4);

  const __bf16* kbb = Kb + (size_t)b * NN * KP;
  const __bf16* vbb = Vt + (size_t)b * CC * NN;

  // resident Q fragments: 16 x 160 = 5 A-fragments per wave
  bfx16 qf[5];
  {
    const __bf16* qrow = Qb + ((size_t)b * NN + n0 + col) * KP + kh * 16;
#pragma unroll
    for (int t = 0; t < 5; ++t) qf[t] = lda_frag(qrow + t * 32);
  }

  fx8 acc[32];
#pragma unroll
  for (int t = 0; t < 32; ++t) acc[t] = fx8{};
  float mrow[8], lrow[8];
#pragma unroll
  for (int j = 0; j < 8; ++j) { mrow[j] = -1e30f; lrow[j] = 0.0f; }

  // ---- stage helper: copy K chunk (32 rows x KP) + V chunk (CC x 32) ----
  auto stage = [&](int buf, int m0) {
    // K rows m0..m0+31 are contiguous: 5120 elems = 640 x 16B
#pragma unroll
    for (int i = 0; i < 5; ++i) {
      int p = tid + i * 128;
      cp16(&ldsK[buf][p * 8], kbb + (size_t)m0 * KP + p * 8);
    }
    // V: 512 channels x 32 m-values: 2048 x 16B
#pragma unroll
    for (int i = 0; i < 16; ++i) {
      int p = tid + i * 128;
      int ch = p >> 2, seg = p & 3;
      cp16(&ldsV[buf][ch * 32 + seg * 8], vbb + (size_t)ch * NN + m0 + seg * 8);
    }
  };

  // prologue: stage chunk 0
  stage(0, 0);
  wait_async_all();
  __syncthreads();

  for (int m0 = 0, it = 0; m0 < NN; m0 += 32, ++it) {
    const int cur = it & 1;
    if (m0 + 32 < NN) stage(cur ^ 1, m0 + 32);   // overlap with compute

    const __bf16* lk = ldsK[cur];
    const __bf16* lv = ldsV[cur];

    // ---- S = Q K^T : two 16x16 tiles
    fx8 s0 = {}, s1 = {};
#pragma unroll
    for (int t = 0; t < 5; ++t) {
      s0 = wmma_bf16(qf[t], ldb_frag(lk + (size_t)col * KP + t * 32 + kh * 16), s0);
      s1 = wmma_bf16(qf[t], ldb_frag(lk + (size_t)(16 + col) * KP + t * 32 + kh * 16), s1);
    }

    // ---- online softmax update
    float scale[8];
#pragma unroll
    for (int j = 0; j < 8; ++j) {
      float mx = redmax16(fmaxf(s0[j], s1[j]));
      float mn = fmaxf(mrow[j], mx);
      scale[j] = __expf(mrow[j] - mn);
      mrow[j] = mn;
      float p0 = __expf(s0[j] - mn);
      float p1 = __expf(s1[j] - mn);
      lrow[j] = lrow[j] * scale[j] + redsum16(p0 + p1);
      const int r = j + kh * 8;
      pbuf[wave][r * 32 + col] = (__bf16)p0;        // intra-wave DS: in-order
      pbuf[wave][r * 32 + 16 + col] = (__bf16)p1;
    }

    // rescale O accumulators
#pragma unroll
    for (int t = 0; t < 32; ++t)
#pragma unroll
      for (int j = 0; j < 8; ++j) acc[t][j] *= scale[j];

    // re-layout P from LDS as bf16 A-fragment (16x32)
    bfx16 pf = lda_frag(&pbuf[wave][col * 32 + kh * 16]);

    // ---- O += P * V over all 512 channels (32 tiles)
#pragma unroll
    for (int t = 0; t < 32; ++t)
      acc[t] = wmma_bf16(pf, ldb_frag(lv + (t * 16 + col) * 32 + kh * 16), acc[t]);

    wait_async_all();   // next chunk staged (this wave's copies)
    __syncthreads();    // all waves done with cur + staging complete
  }

  // ---- normalize and store feats (bf16)
#pragma unroll
  for (int t = 0; t < 32; ++t) {
#pragma unroll
    for (int j = 0; j < 8; ++j) {
      const float o = acc[t][j] / lrow[j];
      Fb[((size_t)b * NN + n0 + j + kh * 8) * CC + t * 16 + col] = (__bf16)o;
    }
  }
}

// ---------------- kernel 4: output projection + gamma residual ----------------
__global__ void out_kernel(const __bf16* __restrict__ Fb,
                           const __bf16* __restrict__ wrb,
                           const float* __restrict__ bres,
                           const float* __restrict__ xf,
                           const float* __restrict__ gamma,
                           float* __restrict__ out2) {
  const int lane = threadIdx.x & 31;
  const int col = lane & 15, kh = lane >> 4;
  const int jbase = blockIdx.x << 4;
  const int rt = blockIdx.y;
  const int b = rt >> 8, n0 = (rt & 255) << 4;

  const __bf16* arow = Fb + ((size_t)(b * NN + n0 + col)) * CC + kh * 16;
  const __bf16* brow = wrb + (size_t)(jbase + col) * CC + kh * 16;

  fx8 acc = {};
#pragma unroll 4
  for (int cc = 0; cc < CC; cc += 32)
    acc = wmma_bf16(lda_frag(arow + cc), ldb_frag(brow + cc), acc);

  const int e = jbase + col;
  const float g = gamma[0];
  const float bval = bres[e];
#pragma unroll
  for (int j = 0; j < 8; ++j) {
    const size_t idx = ((size_t)b * NN + n0 + j + kh * 8) * CC + e;
    out2[idx] = g * (acc[j] + bval) + xf[idx];
  }
}

// ---------------- host side ----------------
extern "C" void kernel_launch(void* const* d_in, const int* in_sizes, int n_in,
                              void* d_out, int out_size, void* d_ws, size_t ws_size,
                              hipStream_t stream) {
  const float* x    = (const float*)d_in[0];
  const float* Wq   = (const float*)d_in[1];
  const float* bq   = (const float*)d_in[2];
  const float* Wk   = (const float*)d_in[3];
  const float* bk   = (const float*)d_in[4];
  const float* Wv   = (const float*)d_in[5];
  const float* bv   = (const float*)d_in[6];
  const float* Wres = (const float*)d_in[7];
  const float* bres = (const float*)d_in[8];
  const float* gam  = (const float*)d_in[9];

  float* segmap  = (float*)d_out;                           // (B,N,K)
  float* featmap = (float*)d_out + (size_t)BN * NN * KK;    // (B,N,C)

  char* ws = (char*)d_ws;
  __bf16* xb  = (__bf16*)(ws + 0);
  __bf16* wqb = (__bf16*)(ws + 33554432);
  __bf16* wkb = (__bf16*)(ws + 33718272);
  __bf16* wvb = (__bf16*)(ws + 33882112);
  __bf16* wrb = (__bf16*)(ws + 34406400);
  __bf16* Qb  = (__bf16*)(ws + 34930688);
  __bf16* Kb  = (__bf16*)(ws + 45416448);
  __bf16* Vt  = (__bf16*)(ws + 55902208);
  __bf16* Fb  = (__bf16*)(ws + 89456640);   // end: 123,011,072 bytes

  const int row_tiles = BN * NN / 16;  // 2048

  cvt_kernel<<<4096, 256, 0, stream>>>(x, Wq, Wk, Wv, Wres, xb, wqb, wkb, wvb, wrb);
  proj_qk_kernel<<<dim3(KP / 16, row_tiles, 2), 32, 0, stream>>>(
      xb, wqb, wkb, bq, bk, segmap, Qb, Kb);
  proj_v_kernel<<<dim3(CC / 16, row_tiles, 1), 32, 0, stream>>>(xb, wvb, bv, Vt);
  attn_kernel<<<dim3(BN * NN / 64, 1, 1), 128, 0, stream>>>(Qb, Kb, Vt, Fb);
  out_kernel<<<dim3(CC / 16, row_tiles, 1), 32, 0, stream>>>(
      Fb, wrb, bres, x, gam, featmap);
}
